// StandardAttention_1348619731142
// MI455X (gfx1250) — matrix operand order repaired and checked by reading, not compiled, not run on hardware
//
#include <hip/hip_runtime.h>
#include <hip/hip_bf16.h>
#include <math.h>

// ---------------------------------------------------------------------------
// Fused attention block for MI455X (gfx1250, wave32, WMMA bf16 16x16x32).
// B=2, T=2048, D=1024, H=16, hd=64.
// Pipeline: f32->bf16 convert -> QKV GEMMs (async-LDS B staging, RoPE fused)
//           -> flash attention (online softmax, all WMMA) -> output projection.
// ---------------------------------------------------------------------------

typedef __bf16 bf16_t;
typedef __attribute__((ext_vector_type(16))) __bf16 v16bf;
typedef __attribute__((ext_vector_type(8)))  float  v8f;

union ABfrag { uint4 u[2]; v16bf v; };

static __device__ inline v8f wmma_bf16(v16bf a, v16bf b, v8f c) {
  // (neg_a, A, neg_b, B, c_mod, C, reuse_a, reuse_b)
  return __builtin_amdgcn_wmma_f32_16x16x32_bf16(false, a, false, b,
                                                 (short)0, c, false, false);
}

// round-to-nearest-even f32 -> bf16
static __device__ inline bf16_t f2bf(float f) {
  unsigned u = __float_as_uint(f);
  unsigned r = (u + 0x7FFFu + ((u >> 16) & 1u)) >> 16;
  unsigned short h = (unsigned short)r;
  return __builtin_bit_cast(bf16_t, h);
}

static __device__ inline float rmax16(float v) {
#pragma unroll
  for (int m = 1; m < 16; m <<= 1) v = fmaxf(v, __shfl_xor(v, m, 32));
  return v;
}
static __device__ inline float rsum16(float v) {
#pragma unroll
  for (int m = 1; m < 16; m <<= 1) v += __shfl_xor(v, m, 32);
  return v;
}

// async copy 16B global -> LDS (ASYNCcnt-tracked, no VGPR round trip)
static __device__ inline void async_copy16(unsigned lds_off,
                                           const bf16_t* gptr) {
  asm volatile("global_load_async_to_lds_b128 %0, %1, off"
               :: "v"(lds_off), "v"(gptr) : "memory");
}
static __device__ inline void wait_async0() {
  asm volatile("s_wait_asynccnt 0x0" ::: "memory");
}

// ---------------------------------------------------------------------------
__global__ void cvt_f32_bf16(const float* __restrict__ s,
                             bf16_t* __restrict__ d, int n) {
  int i = blockIdx.x * blockDim.x + threadIdx.x;
  if (i < n) d[i] = f2bf(s[i]);
}

// ---------------------------------------------------------------------------
// C = A[4096,1024] @ W[1024,1024]^T  (bf16 in, f32 accum)
// Block = 128 threads = 4 waves; all waves share n0 (64 N cols), each wave a
// 16-row M tile. B tile (64x32 bf16 = 4KB) staged in LDS per K-step via
// global_load_async_to_lds_b128 and shared by all 4 waves.
// mode 0: RoPE epilogue -> bf16 [B,H,T,hd]       (Q, K)
// mode 1: transpose epilogue -> bf16 [B,H,hd,T]  (V)
// mode 2: plain f32 [M,N]                        (final @ Wo^T)
#define MODE_ROPE 0
#define MODE_VT   1
#define MODE_F32  2

__global__ __launch_bounds__(128) void gemm16(
    const bf16_t* __restrict__ A, const bf16_t* __restrict__ W,
    bf16_t* __restrict__ outB, float* __restrict__ outF,
    const float* __restrict__ cosT, const float* __restrict__ sinT, int mode) {
  __shared__ __align__(16) bf16_t sB[64 * 32];  // [n_in_tile][k_in_tile]
  const int tid  = threadIdx.x;
  const int lane = tid & 31;
  const int wave = tid >> 5;
  const int lo = lane & 15, hi = lane >> 4;
  const int m0 = blockIdx.x * 64 + wave * 16;
  const int n0 = blockIdx.y * 64;

  const unsigned ldsbase = (unsigned)(size_t)(&sB[0]);  // LDS byte offset
  const int c0 = tid, c1 = tid + 128;  // 16B chunk ids (256 chunks total)
  const bf16_t* Wc0 = W + (size_t)(n0 + (c0 >> 2)) * 1024 + (c0 & 3) * 8;
  const bf16_t* Wc1 = W + (size_t)(n0 + (c1 >> 2)) * 1024 + (c1 & 3) * 8;

  v8f acc[4] = {};
  const bf16_t* Arow = A + (size_t)(m0 + lo) * 1024;

  for (int k0 = 0; k0 < 1024; k0 += 32) {
    // stage 64x32 B tile into LDS (2 async 16B chunks per thread)
    async_copy16(ldsbase + (unsigned)c0 * 16u, Wc0 + k0);
    async_copy16(ldsbase + (unsigned)c1 * 16u, Wc1 + k0);
    wait_async0();
    __syncthreads();

    if (k0 + 32 < 1024) __builtin_prefetch(Arow + k0 + 32, 0, 3);
    // A fragment (16x32, 16-bit): lane lo=row M, chunks at K=hi*8 and 16+hi*8
    ABfrag af;
    af.u[0] = *(const uint4*)(Arow + k0 + hi * 8);
    af.u[1] = *(const uint4*)(Arow + k0 + 16 + hi * 8);
#pragma unroll
    for (int j = 0; j < 4; ++j) {
      // B fragment (32x16): lane col N=lo, chunks at K=hi*8 and 16+hi*8 from LDS
      const bf16_t* Brow = sB + (j * 16 + lo) * 32 + hi * 8;
      ABfrag bfB;
      bfB.u[0] = *(const uint4*)(Brow);
      bfB.u[1] = *(const uint4*)(Brow + 16);
      acc[j] = wmma_bf16(af.v, bfB.v, acc[j]);
    }
    __syncthreads();  // protect sB before next K-step overwrites it
  }

  if (mode == MODE_F32) {
#pragma unroll
    for (int j = 0; j < 4; ++j)
#pragma unroll
      for (int r = 0; r < 8; ++r) {
        int m = m0 + r + 8 * hi;
        int n = n0 + j * 16 + lo;
        outF[(size_t)m * 1024 + n] = acc[j][r];
      }
  } else if (mode == MODE_VT) {
#pragma unroll
    for (int j = 0; j < 4; ++j)
#pragma unroll
      for (int r = 0; r < 8; ++r) {
        int m = m0 + r + 8 * hi;
        int b = m >> 11, t = m & 2047;
        int n = n0 + j * 16 + lo;
        int h = n >> 6, d = n & 63;
        outB[((size_t)(b * 16 + h) * 64 + d) * 2048 + t] = f2bf(acc[j][r]);
      }
  } else {  // MODE_ROPE: pairs (2j,2j+1) are adjacent lanes -> shfl_xor(1)
#pragma unroll
    for (int j = 0; j < 4; ++j)
#pragma unroll
      for (int r = 0; r < 8; ++r) {
        int m = m0 + r + 8 * hi;
        int b = m >> 11, t = m & 2047;
        int n = n0 + j * 16 + lo;
        int h = n >> 6, d = n & 63;
        float v = acc[j][r];
        float p = __shfl_xor(v, 1, 32);  // partner dim value
        int jj = d >> 1;
        float c = cosT[t * 32 + jj];
        float s = sinT[t * 32 + jj];
        float res = ((d & 1) == 0) ? (v * c - p * s) : (p * s + v * c);
        outB[((size_t)(b * 16 + h) * 2048 + t) * 64 + d] = f2bf(res);
      }
  }
}

// ---------------------------------------------------------------------------
// Flash attention: one wave per 16-query tile; 32-key blocks; causal.
// Q,K: [B,H,T,64] bf16.  Vt: [B,H,64,T] bf16.  O: [B,T,1024] bf16.
__global__ __launch_bounds__(128) void flash16(
    const bf16_t* __restrict__ Q, const bf16_t* __restrict__ K,
    const bf16_t* __restrict__ Vt, bf16_t* __restrict__ O) {
  __shared__ __align__(16) bf16_t pbuf[4][16 * 32];  // per-wave P staging
  const int lane = threadIdx.x & 31;
  const int wave = threadIdx.x >> 5;
  const int lo = lane & 15, hi = lane >> 4;
  const int q0 = (blockIdx.x * 4 + wave) * 16;
  const int h = blockIdx.y, b = blockIdx.z;
  const bf16_t* Qh = Q + (size_t)(b * 16 + h) * 2048 * 64;
  const bf16_t* Kh = K + (size_t)(b * 16 + h) * 2048 * 64;
  const bf16_t* Vh = Vt + (size_t)(b * 16 + h) * 64 * 2048;
  bf16_t* pb = pbuf[wave];

  // Q fragments (two K=32 chunks covering hd=64), loaded once
  ABfrag qf[2];
  {
    const bf16_t* Qrow = Qh + (size_t)(q0 + lo) * 64;
    qf[0].u[0] = *(const uint4*)(Qrow + hi * 8);
    qf[0].u[1] = *(const uint4*)(Qrow + 16 + hi * 8);
    qf[1].u[0] = *(const uint4*)(Qrow + 32 + hi * 8);
    qf[1].u[1] = *(const uint4*)(Qrow + 48 + hi * 8);
  }

  float mrow[8], lrow[8];
#pragma unroll
  for (int r = 0; r < 8; ++r) { mrow[r] = -INFINITY; lrow[r] = 0.f; }
  v8f oacc[4] = {};
  const float scale = 0.125f;  // 64^-0.5

  const int kend = q0 + 16;  // causal: keys < q0+16
  for (int k0 = 0; k0 < kend; k0 += 32) {
    // S = Q @ K^T for 32 keys -> two 16x16 f32 tiles
    v8f sfr[2];
#pragma unroll
    for (int half = 0; half < 2; ++half) {
      v8f s = {};
#pragma unroll
      for (int kc = 0; kc < 2; ++kc) {
        const bf16_t* Krow =
            Kh + (size_t)(k0 + half * 16 + lo) * 64 + kc * 32 + hi * 8;
        ABfrag bfB;
        bfB.u[0] = *(const uint4*)(Krow);
        bfB.u[1] = *(const uint4*)(Krow + 16);
        s = wmma_bf16(qf[kc].v, bfB.v, s);
      }
      sfr[half] = s;
    }
    // online softmax per row r (row = q0 + r + 8*hi; col group = lanes lo)
#pragma unroll
    for (int r = 0; r < 8; ++r) {
      int qg = q0 + r + 8 * hi;
      float s0 = sfr[0][r] * scale;
      float s1 = sfr[1][r] * scale;
      if (k0 + lo > qg) s0 = -1e30f;
      if (k0 + 16 + lo > qg) s1 = -1e30f;
      float mx = rmax16(fmaxf(s0, s1));
      float nm = fmaxf(mrow[r], mx);
      float al = __expf(mrow[r] - nm);
      float p0 = __expf(s0 - nm);
      float p1 = __expf(s1 - nm);
      float rs = rsum16(p0 + p1);
      lrow[r] = lrow[r] * al + rs;
      mrow[r] = nm;
#pragma unroll
      for (int j = 0; j < 4; ++j) oacc[j][r] *= al;
      // stage P[m][k] (16x32, row-major) in LDS for C->A layout transpose
      pb[(r + 8 * hi) * 32 + lo] = f2bf(p0);
      pb[(r + 8 * hi) * 32 + 16 + lo] = f2bf(p1);
    }
    // read back as A fragment (DS ops are in-order within a wave)
    ABfrag pf;
    pf.u[0] = *(const uint4*)(pb + lo * 32 + hi * 8);
    pf.u[1] = *(const uint4*)(pb + lo * 32 + 16 + hi * 8);
    // O += P @ V  (V transposed: per-lane contiguous K runs)
#pragma unroll
    for (int j = 0; j < 4; ++j) {
      const bf16_t* Vrow = Vh + (size_t)(j * 16 + lo) * 2048 + k0 + hi * 8;
      ABfrag bv;
      bv.u[0] = *(const uint4*)(Vrow);
      bv.u[1] = *(const uint4*)(Vrow + 16);
      oacc[j] = wmma_bf16(pf.v, bv.v, oacc[j]);
    }
  }

  // normalize and store O as [B,T,(h,hd)] bf16
#pragma unroll
  for (int j = 0; j < 4; ++j)
#pragma unroll
    for (int r = 0; r < 8; ++r) {
      int t = q0 + r + 8 * hi;
      int dd = h * 64 + j * 16 + lo;
      float o = oacc[j][r] / lrow[r];
      O[((size_t)b * 2048 + t) * 1024 + dd] = f2bf(o);
    }
}

// ---------------------------------------------------------------------------
extern "C" void kernel_launch(void* const* d_in, const int* in_sizes, int n_in,
                              void* d_out, int out_size, void* d_ws,
                              size_t ws_size, hipStream_t stream) {
  (void)in_sizes; (void)n_in; (void)out_size; (void)ws_size;
  const float* x  = (const float*)d_in[0];
  const float* Wq = (const float*)d_in[1];
  const float* Wk = (const float*)d_in[2];
  const float* Wv = (const float*)d_in[3];
  const float* Wo = (const float*)d_in[4];
  const float* fc = (const float*)d_in[5];  // [2048,32]
  const float* fs = (const float*)d_in[6];
  float* out = (float*)d_out;

  // workspace carve-up (bf16 elements), total = 48 MB
  bf16_t* xbf = (bf16_t*)d_ws;
  bf16_t* wqb = xbf + (size_t)4096 * 1024;
  bf16_t* wkb = wqb + (size_t)1024 * 1024;
  bf16_t* wvb = wkb + (size_t)1024 * 1024;
  bf16_t* wob = wvb + (size_t)1024 * 1024;
  bf16_t* Qb  = wob + (size_t)1024 * 1024;   // [B,H,T,64]
  bf16_t* Kb  = Qb  + (size_t)4096 * 1024;   // [B,H,T,64]
  bf16_t* Vtb = Kb  + (size_t)4096 * 1024;   // [B,H,64,T]
  bf16_t* Ob  = Vtb + (size_t)4096 * 1024;   // [B,T,1024]

  const int nx = 4096 * 1024, nw = 1024 * 1024;
  cvt_f32_bf16<<<(nx + 255) / 256, 256, 0, stream>>>(x,  xbf, nx);
  cvt_f32_bf16<<<(nw + 255) / 256, 256, 0, stream>>>(Wq, wqb, nw);
  cvt_f32_bf16<<<(nw + 255) / 256, 256, 0, stream>>>(Wk, wkb, nw);
  cvt_f32_bf16<<<(nw + 255) / 256, 256, 0, stream>>>(Wv, wvb, nw);
  cvt_f32_bf16<<<(nw + 255) / 256, 256, 0, stream>>>(Wo, wob, nw);

  dim3 gg(64, 16);  // 4096/64 M-tiles x 1024/64 N-tiles
  gemm16<<<gg, 128, 0, stream>>>(xbf, wqb, Qb,  nullptr, fc, fs, MODE_ROPE);
  gemm16<<<gg, 128, 0, stream>>>(xbf, wkb, Kb,  nullptr, fc, fs, MODE_ROPE);
  gemm16<<<gg, 128, 0, stream>>>(xbf, wvb, Vtb, nullptr, fc, fs, MODE_VT);

  flash16<<<dim3(32, 16, 2), 128, 0, stream>>>(Qb, Kb, Vtb, Ob);

  gemm16<<<gg, 128, 0, stream>>>(Ob, wob, nullptr, out, fc, fs, MODE_F32);
}
